// TransformerGNNIntegration_37864431681840
// MI455X (gfx1250) — matrix operand order repaired and checked by reading, not compiled, not run on hardware
//
#include <hip/hip_runtime.h>

// ---------------------------------------------------------------------------
// Problem constants (from reference)
// ---------------------------------------------------------------------------
constexpr int B  = 4;
constexpr int S  = 2048;
constexpr int H  = 768;
constexpr int G  = 128;
constexpr int NN = 40000;
constexpr int EE = 640000;
constexpr int L  = 3;
constexpr int STEPS = 2;
constexpr int BS = B * S;          // 8192
constexpr float NEG_SLOPE = 0.2f;
constexpr float LN_EPS = 1e-5f;

constexpr int KMAX = 768;          // largest GEMM K (H)
constexpr int KPAD = 8;            // LDS row pad: DWORD stride % 64 == 4 -> no bank conflicts

typedef __bf16 bf16_t;
typedef __attribute__((ext_vector_type(16))) __bf16 v16bf;
typedef __attribute__((ext_vector_type(8)))  __bf16 v8bf;
typedef __attribute__((ext_vector_type(8)))  float  v8f;

// ---------------------------------------------------------------------------
// Helpers
// ---------------------------------------------------------------------------
__device__ __forceinline__ int enc_f(float f) {
    int i = __float_as_int(f);
    return i < 0 ? (i ^ 0x7fffffff) : i;   // order-preserving int encoding
}
__device__ __forceinline__ float dec_f(int i) {
    return __int_as_float(i < 0 ? (i ^ 0x7fffffff) : i);
}

// ---------------------------------------------------------------------------
// Elementwise / utility kernels
// ---------------------------------------------------------------------------
__global__ void k_f32_to_bf16(bf16_t* __restrict__ dst, const float* __restrict__ src, size_t n) {
    size_t i = (size_t)blockIdx.x * blockDim.x + threadIdx.x;
    if (i < n) dst[i] = (bf16_t)src[i];
}

// dst[n*K + k] = (bf16) src[k*N + n]   (transpose [K,N] -> [N,K])
__global__ void k_transpose_bf16(bf16_t* __restrict__ dst, const float* __restrict__ src, int K, int N) {
    size_t i = (size_t)blockIdx.x * blockDim.x + threadIdx.x;
    if (i >= (size_t)K * N) return;
    int n = (int)(i / K);
    int k = (int)(i % K);
    dst[i] = (bf16_t)src[(size_t)k * N + n];
}

__global__ void k_init_f32(float* __restrict__ p, size_t n, float v) {
    size_t i = (size_t)blockIdx.x * blockDim.x + threadIdx.x;
    if (i < n) p[i] = v;
}
__global__ void k_init_i32(int* __restrict__ p, size_t n, int v) {
    size_t i = (size_t)blockIdx.x * blockDim.x + threadIdx.x;
    if (i < n) p[i] = v;
}

// ---------------------------------------------------------------------------
// WMMA bf16 GEMM:  C[M,N](f32) = epilogue( A[M,K](bf16) @ Bt[N,K](bf16) )
// Bt is the weight pre-transposed to [N,K].  Requires M % 32 == 0.
// The 32-column weight strip for the block is staged once into LDS with
// CDNA5 async global->LDS copies (ASYNCcnt), row-padded so each
// ds_load_b128 hits 4 distinct banks per lane (stride % 64 dw == 4).
// Each wave computes a 32x32 output patch: 2 M-tiles x 2 N-tiles = 4 WMMA
// per k-step sharing both B fragments.
// act: 0 = acc + bias            (bias may be null)
//      1 = relu(acc + bias) + C  (in-place residual)
//      2 = tanh(acc)
//      3 = C += acc              (accumulate into existing C)
// ---------------------------------------------------------------------------
__global__ __launch_bounds__(256)
void k_gemm_bf16(const bf16_t* __restrict__ A, const bf16_t* __restrict__ Bt,
                 const float* __restrict__ bias, float* __restrict__ C,
                 int M, int N, int K, int act)
{
    __shared__ bf16_t Bs[32 * (KMAX + KPAD)];   // 49,664 B
    const int Kp   = K + KPAD;
    const int lane = threadIdx.x & 31;
    const int wave = threadIdx.x >> 5;
    const int n0   = blockIdx.x * 32;

    // ---- async-stage B strip [n0..n0+32) x K into LDS (row stride Kp) ----
    {
        const int rowBytes     = K * 2;
        const int chunksPerRow = rowBytes >> 4;          // 16-byte chunks
        const int totalChunks  = 32 * chunksPerRow;
        const char* gbase = (const char*)(Bt + (size_t)n0 * K);
        for (int c = threadIdx.x; c < totalChunks; c += 256) {
            int r  = c / chunksPerRow;
            int cc = c - r * chunksPerRow;
            unsigned lds_off = (unsigned)(r * Kp * 2 + cc * 16);
            const char* gp   = gbase + (size_t)r * rowBytes + (size_t)cc * 16;
            asm volatile("global_load_async_to_lds_b128 %0, %1, off"
                         :: "v"(lds_off), "v"(gp) : "memory");
        }
        // Pass Bs into the asm so the LDS object escapes: the memory clobber
        // then forces the compiler to treat Bs as written by the async copies
        // (otherwise the ds_loads below fold to undef and vanish).
        asm volatile("s_wait_asynccnt 0" :: "v"(Bs) : "memory");
        __syncthreads();
    }

    // two 16-row tiles per wave (M is always a multiple of 32 here)
    const int m0 = (blockIdx.y * 8 + wave) * 32;
    if (m0 >= M) return;                 // wave-uniform; after the only barrier
    const int m1 = m0 + 16;

    // A fragment addressing (16-bit A 16x32 layout, ISA 7.12.2)
    const int am  = lane & 15;
    const int akb = (lane >> 4) << 3;    // 0 or 8
    const bf16_t* pa0 = A + (size_t)(m0 + am) * K + akb;
    const bf16_t* pa1 = A + (size_t)(m1 + am) * K + akb;

    // B fragments from LDS: lane holds column, K = kb..kb+7 and kb+16..kb+23
    const int bn  = lane & 15;
    const int bkb = (lane >> 4) << 3;    // 0 or 8
    const bf16_t* pb0 = Bs + (size_t)bn * Kp + bkb;
    const bf16_t* pb1 = Bs + (size_t)(16 + bn) * Kp + bkb;

    v8f acc00 = {}, acc01 = {}, acc10 = {}, acc11 = {};
    for (int k0 = 0; k0 < K; k0 += 32) {
        v8bf a0lo = *(const v8bf*)(pa0 + k0);
        v8bf a0hi = *(const v8bf*)(pa0 + k0 + 16);
        v8bf a1lo = *(const v8bf*)(pa1 + k0);
        v8bf a1hi = *(const v8bf*)(pa1 + k0 + 16);
        v16bf a0, a1;
        #pragma unroll
        for (int i = 0; i < 8; ++i) {
            a0[i] = a0lo[i]; a0[i + 8] = a0hi[i];
            a1[i] = a1lo[i]; a1[i + 8] = a1hi[i];
        }
        v8bf b0lo = *(const v8bf*)(pb0 + k0);
        v8bf b0hi = *(const v8bf*)(pb0 + k0 + 16);
        v8bf b1lo = *(const v8bf*)(pb1 + k0);
        v8bf b1hi = *(const v8bf*)(pb1 + k0 + 16);
        v16bf b0, b1;
        #pragma unroll
        for (int i = 0; i < 8; ++i) {
            b0[i] = b0lo[i]; b0[i + 8] = b0hi[i];
            b1[i] = b1lo[i]; b1[i + 8] = b1hi[i];
        }
        // interleave independent WMMAs to minimise hazard NOPs
        acc00 = __builtin_amdgcn_wmma_f32_16x16x32_bf16(false, a0, false, b0,
                                                        (short)0, acc00, false, false);
        acc10 = __builtin_amdgcn_wmma_f32_16x16x32_bf16(false, a1, false, b0,
                                                        (short)0, acc10, false, false);
        acc01 = __builtin_amdgcn_wmma_f32_16x16x32_bf16(false, a0, false, b1,
                                                        (short)0, acc01, false, false);
        acc11 = __builtin_amdgcn_wmma_f32_16x16x32_bf16(false, a1, false, b1,
                                                        (short)0, acc11, false, false);
    }

    // C/D layout: VGPR r: lanes 0-15 -> M=mb+r, lanes 16-31 -> M=mb+8+r; N=n0+(lane&15)
    const int cn = n0 + (lane & 15);
    float b0v = 0.f, b1v = 0.f;
    if ((act == 0 && bias) || act == 1) { b0v = bias[cn]; b1v = bias[cn + 16]; }

    auto epi = [&](int mb, const v8f& e0, const v8f& e1) {
        const int cmb = mb + ((lane >> 4) << 3);
        #pragma unroll
        for (int r = 0; r < 8; ++r) {
            size_t i0 = (size_t)(cmb + r) * N + cn;
            float v0 = e0[r], v1 = e1[r];
            if (act == 0) {
                C[i0]      = v0 + b0v;
                C[i0 + 16] = v1 + b1v;
            } else if (act == 1) {
                C[i0]      = fmaxf(v0 + b0v, 0.f) + C[i0];
                C[i0 + 16] = fmaxf(v1 + b1v, 0.f) + C[i0 + 16];
            } else if (act == 2) {
                C[i0]      = tanhf(v0);
                C[i0 + 16] = tanhf(v1);
            } else {
                C[i0]      += v0;
                C[i0 + 16] += v1;
            }
        }
    };
    epi(m0, acc00, acc01);
    epi(m1, acc10, acc11);
}

// ---------------------------------------------------------------------------
// Gather node rows: h[n,:] = h_seq[bidx[n]*S + nidx[n], :]
// ---------------------------------------------------------------------------
__global__ void k_gather(float* __restrict__ h, const float* __restrict__ h_seq,
                         const int* __restrict__ nidx, const int* __restrict__ bidx) {
    int i = blockIdx.x * blockDim.x + threadIdx.x;
    if (i >= NN * G) return;
    int n = i >> 7, g = i & 127;
    h[i] = h_seq[((size_t)bidx[n] * S + nidx[n]) * G + g];
}

// wave-per-node dual dot: s1[n] = m[n,:].a1 ; s2[n] = m[n,:].a2
__global__ void k_rowdot2(float* __restrict__ s1, float* __restrict__ s2,
                          const float* __restrict__ m,
                          const float* __restrict__ a1, const float* __restrict__ a2) {
    int wid  = (blockIdx.x * blockDim.x + threadIdx.x) >> 5;
    int lane = threadIdx.x & 31;
    if (wid >= NN) return;
    const float* row = m + (size_t)wid * G;
    float x1 = 0.f, x2 = 0.f;
    for (int g = lane; g < G; g += 32) { float v = row[g]; x1 += v * a1[g]; x2 += v * a2[g]; }
    for (int o = 16; o; o >>= 1) { x1 += __shfl_xor(x1, o, 32); x2 += __shfl_xor(x2, o, 32); }
    if (lane == 0) { s1[wid] = x1; s2[wid] = x2; }
}

__global__ void k_rowdot1(float* __restrict__ s1, const float* __restrict__ m,
                          const float* __restrict__ a1) {
    int wid  = (blockIdx.x * blockDim.x + threadIdx.x) >> 5;
    int lane = threadIdx.x & 31;
    if (wid >= NN) return;
    const float* row = m + (size_t)wid * G;
    float x1 = 0.f;
    for (int g = lane; g < G; g += 32) x1 += row[g] * a1[g];
    for (int o = 16; o; o >>= 1) x1 += __shfl_xor(x1, o, 32);
    if (lane == 0) s1[wid] = x1;
}

// ---------------------------------------------------------------------------
// Edge kernels (concatenated src/dst: e<EE -> (esrc,edst) else flipped)
// ---------------------------------------------------------------------------
__global__ void k_edge_score(float* __restrict__ score, int* __restrict__ smax,
                             const float* __restrict__ ssrc, const float* __restrict__ sdst,
                             const int* __restrict__ esrc, const int* __restrict__ edst) {
    int e = blockIdx.x * blockDim.x + threadIdx.x;
    if (e >= 2 * EE) return;
    int s = e < EE ? esrc[e] : edst[e - EE];
    int d = e < EE ? edst[e] : esrc[e - EE];
    float x = ssrc[s] + sdst[d];
    x = x > 0.f ? x : NEG_SLOPE * x;
    score[e] = x;
    atomicMax(smax + d, enc_f(x));
}

__global__ void k_edge_exp(float* __restrict__ score, float* __restrict__ denom,
                           const int* __restrict__ smax,
                           const int* __restrict__ esrc, const int* __restrict__ edst) {
    int e = blockIdx.x * blockDim.x + threadIdx.x;
    if (e >= 2 * EE) return;
    int d = e < EE ? edst[e] : esrc[e - EE];
    float sm = dec_f(smax[d]);
    if (sm < -1e37f) sm = 0.f;                 // isfinite() guard from reference
    float ex = expf(score[e] - sm);
    score[e] = ex;                             // in place: score -> exp
    atomicAdd(denom + d, ex);
}

__global__ void k_edge_scatter(float* __restrict__ agg, const float* __restrict__ m,
                               const float* __restrict__ ex, const float* __restrict__ denom,
                               const int* __restrict__ esrc, const int* __restrict__ edst) {
    long long idx = (long long)blockIdx.x * blockDim.x + threadIdx.x;
    if (idx >= (long long)2 * EE * 32) return;
    int e    = (int)(idx >> 5);
    int lane = (int)(idx & 31);
    int s = e < EE ? esrc[e] : edst[e - EE];
    int d = e < EE ? edst[e] : esrc[e - EE];
    float alpha = ex[e] / (denom[d] + 1e-9f);
    const float4 mv = *(const float4*)(m + (size_t)s * G + lane * 4);
    float* o = agg + (size_t)d * G + lane * 4;
    atomicAdd(o + 0, mv.x * alpha);
    atomicAdd(o + 1, mv.y * alpha);
    atomicAdd(o + 2, mv.z * alpha);
    atomicAdd(o + 3, mv.w * alpha);
}

// ---------------------------------------------------------------------------
// Graph readout softmax (B=4 segments)
// ---------------------------------------------------------------------------
__global__ void k_att_max(int* __restrict__ gmax, const float* __restrict__ att,
                          const int* __restrict__ bidx) {
    int n = blockIdx.x * blockDim.x + threadIdx.x;
    if (n >= NN) return;
    atomicMax(gmax + bidx[n], enc_f(att[n]));
}
__global__ void k_att_exp(float* __restrict__ att, float* __restrict__ gden,
                          const int* __restrict__ gmax, const int* __restrict__ bidx) {
    int n = blockIdx.x * blockDim.x + threadIdx.x;
    if (n >= NN) return;
    int b = bidx[n];
    float sm = dec_f(gmax[b]);
    if (sm < -1e37f) sm = 0.f;
    float ex = expf(att[n] - sm);
    att[n] = ex;
    atomicAdd(gden + b, ex);
}

// graph_emb[b,g] = sum_n (bidx[n]==b) h[n,g]*attex[n]/(gden[b]+1e-9)
__global__ __launch_bounds__(128)
void k_graph_emb(float* __restrict__ gemb, const float* __restrict__ h,
                 const float* __restrict__ attex, const float* __restrict__ gden,
                 const int* __restrict__ bidx) {
    int g = threadIdx.x;                // 0..127
    float a0 = 0.f, a1 = 0.f, a2 = 0.f, a3 = 0.f;
    for (int n = blockIdx.x; n < NN; n += gridDim.x) {
        int b = bidx[n];
        float v = h[(size_t)n * G + g] * attex[n] / (gden[b] + 1e-9f);
        a0 += (b == 0) ? v : 0.f;
        a1 += (b == 1) ? v : 0.f;
        a2 += (b == 2) ? v : 0.f;
        a3 += (b == 3) ? v : 0.f;
    }
    atomicAdd(&gemb[0 * G + g], a0);
    atomicAdd(&gemb[1 * G + g], a1);
    atomicAdd(&gemb[2 * G + g], a2);
    atomicAdd(&gemb[3 * G + g], a3);
}

__global__ void k_seq_scatter(float* __restrict__ seqg, const float* __restrict__ h,
                              const int* __restrict__ nidx, const int* __restrict__ bidx) {
    int i = blockIdx.x * blockDim.x + threadIdx.x;
    if (i >= NN * G) return;
    int n = i >> 7, g = i & 127;
    atomicAdd(&seqg[((size_t)bidx[n] * S + nidx[n]) * G + g], h[i]);
}

// A2_bf[b,s,g] = bf16( seqg[b,s,g] + gemb[b,g] )
__global__ void k_add_emb_bf(bf16_t* __restrict__ dst, const float* __restrict__ seqg,
                             const float* __restrict__ gemb) {
    int i = blockIdx.x * blockDim.x + threadIdx.x;
    if (i >= BS * G) return;
    int b = i / (S * G);
    int g = i & 127;
    dst[i] = (bf16_t)(seqg[i] + gemb[b * G + g]);
}

// ---------------------------------------------------------------------------
// Gated residual fusion + LayerNorm. One 256-thread block per row (H=768 = 3*256)
// ---------------------------------------------------------------------------
__global__ __launch_bounds__(256)
void k_fuse_ln(float* __restrict__ out, const float* __restrict__ hid,
               const float* __restrict__ gh, const float* __restrict__ glog,
               const float* __restrict__ gamma, const float* __restrict__ beta) {
    int row = blockIdx.x, t = threadIdx.x;
    __shared__ float s1[256], s2[256];
    float f[3];
    float ls = 0.f, lq = 0.f;
    #pragma unroll
    for (int j = 0; j < 3; ++j) {
        int c = t + j * 256;
        size_t idx = (size_t)row * H + c;
        float gate = 1.f / (1.f + expf(-glog[idx]));
        float v = hid[idx] + gate * gh[idx];
        f[j] = v; ls += v; lq += v * v;
    }
    s1[t] = ls; s2[t] = lq; __syncthreads();
    for (int o = 128; o; o >>= 1) {
        if (t < o) { s1[t] += s1[t + o]; s2[t] += s2[t + o]; }
        __syncthreads();
    }
    float mu   = s1[0] / (float)H;
    float var  = s2[0] / (float)H - mu * mu;
    float rstd = rsqrtf(var + LN_EPS);
    #pragma unroll
    for (int j = 0; j < 3; ++j) {
        int c = t + j * 256;
        size_t idx = (size_t)row * H + c;
        out[idx] = (f[j] - mu) * rstd * gamma[c] + beta[c];
    }
}

// ---------------------------------------------------------------------------
// Host-side launch
// ---------------------------------------------------------------------------
extern "C" void kernel_launch(void* const* d_in, const int* in_sizes, int n_in,
                              void* d_out, int out_size, void* d_ws, size_t ws_size,
                              hipStream_t stream) {
    const float* hidden = (const float*)d_in[0];
    const float* W_in   = (const float*)d_in[1];
    const float* b_in   = (const float*)d_in[2];
    const float* W_msg  = (const float*)d_in[3];
    const float* a_src  = (const float*)d_in[4];
    const float* a_dst  = (const float*)d_in[5];
    const float* W_upd  = (const float*)d_in[6];
    const float* b_upd  = (const float*)d_in[7];
    const float* W_r    = (const float*)d_in[8];
    const float* v_r    = (const float*)d_in[9];
    const float* W_out  = (const float*)d_in[10];
    const float* b_out  = (const float*)d_in[11];
    const float* W_g    = (const float*)d_in[12];
    const float* b_g    = (const float*)d_in[13];
    const float* gamma  = (const float*)d_in[14];
    const float* beta   = (const float*)d_in[15];
    const int* nidx = (const int*)d_in[16];
    const int* bidx = (const int*)d_in[17];
    const int* esrc = (const int*)d_in[18];
    const int* edst = (const int*)d_in[19];
    float* out = (float*)d_out;

    // ---- workspace bump allocator (~152 MB) ----
    char* w = (char*)d_ws;
    auto alloc = [&](size_t bytes) -> char* {
        char* p = w; w += (bytes + 255) & ~size_t(255); return p;
    };
    bf16_t* hid_bf  = (bf16_t*)alloc((size_t)BS * H * 2);
    bf16_t* Win_t   = (bf16_t*)alloc((size_t)G * H * 2);
    bf16_t* Wmsg_t  = (bf16_t*)alloc((size_t)L * G * G * 2);
    bf16_t* Wupd_t  = (bf16_t*)alloc((size_t)L * G * G * 2);
    bf16_t* Wr_t    = (bf16_t*)alloc((size_t)G * G * 2);
    bf16_t* Wout_t  = (bf16_t*)alloc((size_t)H * G * 2);
    bf16_t* Wg1_t   = (bf16_t*)alloc((size_t)H * H * 2);
    bf16_t* Wg2_t   = (bf16_t*)alloc((size_t)H * H * 2);
    float*  h_seq   = (float*)alloc((size_t)BS * G * 4);
    float*  h       = (float*)alloc((size_t)NN * G * 4);
    bf16_t* h_bf    = (bf16_t*)alloc((size_t)NN * G * 2);
    float*  m       = (float*)alloc((size_t)NN * G * 4);   // also tanh buf; glogit aliases
    float*  agg     = (float*)alloc((size_t)NN * G * 4);   // contiguous after m
    bf16_t* agg_bf  = (bf16_t*)alloc((size_t)NN * G * 2);
    float*  s_src   = (float*)alloc((size_t)NN * 4);
    float*  s_dst   = (float*)alloc((size_t)NN * 4);
    int*    smax    = (int*)alloc((size_t)NN * 4);
    float*  denom   = (float*)alloc((size_t)NN * 4);
    float*  score   = (float*)alloc((size_t)2 * EE * 4);
    float*  att     = (float*)alloc((size_t)NN * 4);
    int*    gmax    = (int*)alloc(4 * 4);
    float*  gden    = (float*)alloc(4 * 4);
    float*  gemb    = (float*)alloc((size_t)B * G * 4);
    float*  seqg    = (float*)alloc((size_t)BS * G * 4);
    bf16_t* A2_bf   = (bf16_t*)alloc((size_t)BS * G * 2);
    float*  ghid    = (float*)alloc((size_t)BS * H * 4);
    bf16_t* gh_bf   = (bf16_t*)alloc((size_t)BS * H * 2);
    float*  glogit  = m;     // alias over dead m+agg region (25.2MB <= 41MB)

    auto blk = [](size_t n) { return (unsigned)((n + 255) / 256); };
    auto gemm = [&](const bf16_t* A, const bf16_t* Bt, const float* bias, float* C,
                    int M, int N, int K, int act) {
        dim3 grid(N / 32, (unsigned)((M / 32 + 7) / 8));   // wave = 32 rows x 32 cols
        k_gemm_bf16<<<grid, 256, 0, stream>>>(A, Bt, bias, C, M, N, K, act);
    };

    const float ninf = -__builtin_huge_valf();
    int ninf_i; { union { float f; int i; } u; u.f = ninf; ninf_i = u.i < 0 ? (u.i ^ 0x7fffffff) : u.i; }
    (void)ninf; (void)in_sizes; (void)n_in; (void)out_size; (void)ws_size;

    // ---- prep: bf16 conversions + weight transposes ----
    k_f32_to_bf16<<<blk((size_t)BS * H), 256, 0, stream>>>(hid_bf, hidden, (size_t)BS * H);
    k_transpose_bf16<<<blk((size_t)H * G), 256, 0, stream>>>(Win_t, W_in, H, G);
    for (int l = 0; l < L; ++l) {
        k_transpose_bf16<<<blk((size_t)G * G), 256, 0, stream>>>(Wmsg_t + (size_t)l * G * G, W_msg + (size_t)l * G * G, G, G);
        k_transpose_bf16<<<blk((size_t)G * G), 256, 0, stream>>>(Wupd_t + (size_t)l * G * G, W_upd + (size_t)l * G * G, G, G);
    }
    k_transpose_bf16<<<blk((size_t)G * G), 256, 0, stream>>>(Wr_t, W_r, G, G);
    k_transpose_bf16<<<blk((size_t)G * H), 256, 0, stream>>>(Wout_t, W_out, G, H);
    k_transpose_bf16<<<blk((size_t)H * H), 256, 0, stream>>>(Wg1_t, W_g, H, H);
    k_transpose_bf16<<<blk((size_t)H * H), 256, 0, stream>>>(Wg2_t, W_g + (size_t)H * H, H, H);

    // ---- input projection + gather ----
    gemm(hid_bf, Win_t, b_in, h_seq, BS, G, H, 0);
    k_gather<<<blk((size_t)NN * G), 256, 0, stream>>>(h, h_seq, nidx, bidx);

    // ---- message passing ----
    for (int l = 0; l < L; ++l) {
        for (int st = 0; st < STEPS; ++st) {
            k_f32_to_bf16<<<blk((size_t)NN * G), 256, 0, stream>>>(h_bf, h, (size_t)NN * G);
            gemm(h_bf, Wmsg_t + (size_t)l * G * G, nullptr, m, NN, G, G, 0);
            k_rowdot2<<<blk((size_t)NN * 32), 256, 0, stream>>>(s_src, s_dst, m,
                                                                a_src + (size_t)l * G, a_dst + (size_t)l * G);
            k_init_i32<<<blk(NN), 256, 0, stream>>>(smax, NN, ninf_i);
            k_init_f32<<<blk(NN), 256, 0, stream>>>(denom, NN, 0.f);
            k_init_f32<<<blk((size_t)NN * G), 256, 0, stream>>>(agg, (size_t)NN * G, 0.f);
            k_edge_score<<<blk((size_t)2 * EE), 256, 0, stream>>>(score, smax, s_src, s_dst, esrc, edst);
            k_edge_exp<<<blk((size_t)2 * EE), 256, 0, stream>>>(score, denom, smax, esrc, edst);
            k_edge_scatter<<<blk((size_t)2 * EE * 32), 256, 0, stream>>>(agg, m, score, denom, esrc, edst);
            k_f32_to_bf16<<<blk((size_t)NN * G), 256, 0, stream>>>(agg_bf, agg, (size_t)NN * G);
            gemm(agg_bf, Wupd_t + (size_t)l * G * G, b_upd + (size_t)l * G, h, NN, G, G, 1);
        }
    }

    // ---- attention readout ----
    k_f32_to_bf16<<<blk((size_t)NN * G), 256, 0, stream>>>(h_bf, h, (size_t)NN * G);
    gemm(h_bf, Wr_t, nullptr, m, NN, G, G, 2);                  // m = tanh(h @ W_r)
    k_rowdot1<<<blk((size_t)NN * 32), 256, 0, stream>>>(att, m, v_r);
    k_init_i32<<<1, 4, 0, stream>>>(gmax, 4, ninf_i);
    k_init_f32<<<1, 4, 0, stream>>>(gden, 4, 0.f);
    k_init_f32<<<blk((size_t)B * G), 256, 0, stream>>>(gemb, (size_t)B * G, 0.f);
    k_att_max<<<blk(NN), 256, 0, stream>>>(gmax, att, bidx);
    k_att_exp<<<blk(NN), 256, 0, stream>>>(att, gden, gmax, bidx);
    k_graph_emb<<<1024, 128, 0, stream>>>(gemb, h, att, gden, bidx);

    // ---- scatter back + output projection ----
    k_init_f32<<<blk((size_t)BS * G), 256, 0, stream>>>(seqg, (size_t)BS * G, 0.f);
    k_seq_scatter<<<blk((size_t)NN * G), 256, 0, stream>>>(seqg, h, nidx, bidx);
    k_add_emb_bf<<<blk((size_t)BS * G), 256, 0, stream>>>(A2_bf, seqg, gemb);
    gemm(A2_bf, Wout_t, b_out, ghid, BS, H, G, 0);

    // ---- gate GEMMs (glogit aliases dead m/agg region) ----
    k_f32_to_bf16<<<blk((size_t)BS * H), 256, 0, stream>>>(gh_bf, ghid, (size_t)BS * H);
    gemm(hid_bf, Wg1_t, b_g, glogit, BS, H, H, 0);
    gemm(gh_bf, Wg2_t, nullptr, glogit, BS, H, H, 3);

    // ---- fuse + layernorm ----
    k_fuse_ln<<<BS, 256, 0, stream>>>(out, hidden, ghid, glogit, gamma, beta);
}
